// RPN_19430432047565
// MI455X (gfx1250) — matrix operand order repaired and checked by reading, not compiled, not run on hardware
//
#include <hip/hip_runtime.h>
#include <hip/hip_bf16.h>

// ---------------------------------------------------------------------------
// CDNA5 (gfx1250) RPN pipeline.
// Heavy op: 3x3 conv 512->512 over 64x64 as implicit GEMM on
// v_wmma_f32_16x16x32_bf16, with async global->LDS staging (ASYNCcnt) for
// workgroup-level A/B reuse (64 ochan x 64 pixel macro tile per block).
// ---------------------------------------------------------------------------

typedef __attribute__((ext_vector_type(16))) __bf16 v16bf;
typedef __attribute__((ext_vector_type(8)))  float  v8f;
typedef __attribute__((ext_vector_type(2)))  int    v2i_t;
typedef __attribute__((ext_vector_type(4)))  int    v4i_t;

#define H_IMG 64
#define W_IMG 64
#define C_IN  512
#define NPIX  4096        // 64*64
#define A_NUM 9
#define NANCH 36864       // NPIX * A_NUM
#define POST_NMS 300
#define POOL 7
#define MAXK 12

#if __has_builtin(__builtin_amdgcn_global_load_async_to_lds_b64) && \
    __has_builtin(__builtin_amdgcn_global_load_async_to_lds_b128)
#define HAS_ASYNC_LDS 1
#endif

#ifdef HAS_ASYNC_LDS
#define AS_GLOBAL __attribute__((address_space(1)))
#define AS_LOCAL  __attribute__((address_space(3)))
#endif

__device__ __forceinline__ unsigned short f32_to_bf16(float f) {
    unsigned int u = __float_as_uint(f);
    unsigned int lsb = (u >> 16) & 1u;
    u += 0x7fffu + lsb;                 // round-to-nearest-even
    return (unsigned short)(u >> 16);
}

// K index inside a 32-wide chunk in the contiguous k order this code uses for both A and B:
// lane<16: e0..15 -> K0..15
// lane>=16: e0..15 -> K16..31
__device__ __forceinline__ int a_layout_k(int lane, int e) {
    return ((lane >> 4) << 4) + e;
}

// async copy helpers: 8B / 16B global -> LDS
__device__ __forceinline__ void cp_async_b64(const unsigned short* g, unsigned short* l) {
#ifdef HAS_ASYNC_LDS
    __builtin_amdgcn_global_load_async_to_lds_b64(
        (AS_GLOBAL v2i_t*)g, (AS_LOCAL v2i_t*)l, 0, 0);
#else
    *(unsigned long long*)l = *(const unsigned long long*)g;
#endif
}
__device__ __forceinline__ void cp_async_b128(const unsigned short* g, unsigned short* l) {
#ifdef HAS_ASYNC_LDS
    __builtin_amdgcn_global_load_async_to_lds_b128(
        (AS_GLOBAL v4i_t*)g, (AS_LOCAL v4i_t*)l, 0, 0);
#else
    *(uint4*)l = *(const uint4*)g;
#endif
}
__device__ __forceinline__ void wait_async_lds() {
#ifdef HAS_ASYNC_LDS
#if __has_builtin(__builtin_amdgcn_s_wait_asynccnt)
    __builtin_amdgcn_s_wait_asynccnt(0);
#else
    asm volatile("s_wait_asynccnt 0" ::: "memory");
#endif
#endif
}

// -------------------- conversion / repack kernels --------------------------

// features f32 NCHW -> bf16 padded NHWC (66 x 66 x 512), interior only.
__global__ void k_feat_to_bf16(const float* __restrict__ feat,
                               unsigned short* __restrict__ xp) {
    int idx = blockIdx.x * blockDim.x + threadIdx.x;
    if (idx >= C_IN * NPIX) return;
    int c   = idx >> 12;          // /4096
    int rem = idx & 4095;
    int y   = rem >> 6;
    int x   = rem & 63;
    xp[((y + 1) * 66 + (x + 1)) * C_IN + c] = f32_to_bf16(feat[idx]);
}

// rpn_w (512,512,3,3) f32 -> packed bf16: [(tile_o*9+tap)*16+kc][lane][e]
__global__ void k_repack_conv_w(const float* __restrict__ w,
                                unsigned short* __restrict__ wp) {
    int idx = blockIdx.x * blockDim.x + threadIdx.x;
    const int TOT = 32 * 9 * 16 * 32 * 16;
    if (idx >= TOT) return;
    int e      = idx & 15;
    int lane   = (idx >> 4) & 31;
    int kc     = (idx >> 9) & 15;
    int t      = idx >> 13;
    int tap    = t % 9;
    int tile_o = t / 9;
    int o = tile_o * 16 + (lane & 15);
    int c = kc * 32 + a_layout_k(lane, e);
    int kh = tap / 3, kw = tap % 3;
    wp[idx] = f32_to_bf16(w[((o * C_IN + c) * 3 + kh) * 3 + kw]);
}

// 1x1 weights (odim,512) f32 -> packed bf16 [(otile*16+kc)][lane][e], zero-pad o.
__global__ void k_repack_1x1_w(const float* __restrict__ w, int odim, int n_otiles,
                               unsigned short* __restrict__ wp) {
    int idx = blockIdx.x * blockDim.x + threadIdx.x;
    int tot = n_otiles * 16 * 32 * 16;
    if (idx >= tot) return;
    int e     = idx & 15;
    int lane  = (idx >> 4) & 31;
    int kc    = (idx >> 9) & 15;
    int otile = idx >> 13;
    int o = otile * 16 + (lane & 15);
    int c = kc * 32 + a_layout_k(lane, e);
    float v = (o < odim) ? w[o * C_IN + c] : 0.0f;
    wp[idx] = f32_to_bf16(v);
}

// ----------------------------- WMMA conv -----------------------------------

// 3x3 conv implicit GEMM with LDS staging.
// Block: 512 threads = 16 waves = 4 (o-subtile) x 4 (pixel-subtile).
// Macro tile: 64 out-channels x 64 pixels (one full image row).
// Grid: 8 o-macros * 64 rows = 512 blocks.
#define LDSA_ELEMS (4 * 9 * 512)    // 18432 bf16 = 36 KB
#define LDSB_ELEMS (3 * 66 * 32)    // 6336  bf16 = 12.4 KB

__global__ void __launch_bounds__(512)
k_conv3x3_wmma(const unsigned short* __restrict__ xp,
               const unsigned short* __restrict__ wp,
               const float* __restrict__ bias,
               unsigned short* __restrict__ xo) {
    __shared__ unsigned short lds_a[LDSA_ELEMS];
    __shared__ unsigned short lds_b[LDSB_ELEMS];

    int tid  = threadIdx.x;
    int wave = tid >> 5;
    int lane = tid & 31;
    int wo = wave >> 2;         // o sub-tile   0..3
    int wpx = wave & 3;         // pixel sub-tile 0..3
    int ob = blockIdx.x >> 6;   // o macro 0..7
    int y  = blockIdx.x & 63;   // image row

    int col   = lane & 15;
    int khalf = lane >> 4;

    v8f acc = {};

    for (int kc = 0; kc < 16; ++kc) {
        // ---- stage A: 36 chunks of 1KB (4 o-tiles x 9 taps), b128 units ----
        // unit u: chunk = u>>6 (j*9+tap), within = (u&63)*8 elements
        for (int u = tid; u < 2304; u += 512) {
            int chunk  = u >> 6;
            int within = (u & 63) << 3;
            int j   = chunk / 9;
            int tap = chunk % 9;
            const unsigned short* g =
                wp + ((size_t)(((ob * 4 + j) * 9 + tap) * 16 + kc)) * 512 + within;
            cp_async_b128(g, &lds_a[chunk * 512 + within]);
        }
        // ---- stage B: 3 padded rows x 66 px x 32 ch, b64 units (4 ch) ------
        for (int u = tid; u < 1584; u += 512) {
            int c4    = (u & 7) << 2;
            int pxi   = (u >> 3) % 66;
            int dyr   = (u >> 3) / 66;
            const unsigned short* g =
                xp + ((size_t)((y + dyr) * 66 + pxi)) * C_IN + kc * 32 + c4;
            cp_async_b64(g, &lds_b[(dyr * 66 + pxi) * 32 + c4]);
        }
        wait_async_lds();
        __syncthreads();

        // ---- 9 taps of WMMA from LDS ---------------------------------------
        const unsigned short* abase = &lds_a[(wo * 9) * 512 + lane * 16];
        int bpx = wpx * 16 + col;   // 0..63
        for (int tap = 0; tap < 9; ++tap) {
            v16bf a = *(const v16bf*)(abase + tap * 512);
            const unsigned short* bptr =
                &lds_b[((tap / 3) * 66 + bpx + (tap % 3)) * 32 + khalf * 16];
            v16bf b = *(const v16bf*)bptr;
            acc = __builtin_amdgcn_wmma_f32_16x16x32_bf16(
                false, a, false, b, (short)0, acc, false, false);
        }
        __syncthreads();
    }

    // epilogue: bias + relu, store bf16 [pixel][channel]
    int pix   = y * 64 + wpx * 16 + col;
    int obase = (ob * 4 + wo) * 16 + (khalf ? 8 : 0);
    for (int r = 0; r < 8; ++r) {
        int o = obase + r;
        float v = acc[r] + bias[o];
        v = v > 0.0f ? v : 0.0f;
        xo[(size_t)pix * C_IN + o] = f32_to_bf16(v);
    }
}

// 1x1 conv GEMM on WMMA: out[o][p] = sum_c W[o][c] x[p][c] + b[o]
// grid: n_otiles * 256 waves.
__global__ void k_gemm1x1_wmma(const unsigned short* __restrict__ xo,
                               const unsigned short* __restrict__ wp,
                               const float* __restrict__ bias, int odim,
                               float* __restrict__ out) {
    int wave = (blockIdx.x * blockDim.x + threadIdx.x) >> 5;
    int lane = threadIdx.x & 31;
    int tile_p = wave & 255;
    int otile  = wave >> 8;
    int pbase  = tile_p << 4;
    int col    = lane & 15;
    int khalf  = lane >> 4;

    const unsigned short* xrow = xo + (size_t)(pbase + col) * C_IN + khalf * 16;
    const unsigned short* wrow = wp + (size_t)(otile * 16) * 512 + lane * 16;

    v8f acc = {};
    for (int kc = 0; kc < 16; ++kc) {
        v16bf a = *(const v16bf*)(wrow + kc * 512);
        v16bf b = *(const v16bf*)(xrow + kc * 32);
        acc = __builtin_amdgcn_wmma_f32_16x16x32_bf16(
            false, a, false, b, (short)0, acc, false, false);
    }
    int pix = pbase + col;
    int obase = otile * 16 + (khalf ? 8 : 0);
    for (int r = 0; r < 8; ++r) {
        int o = obase + r;
        float bv = (o < odim) ? bias[o] : 0.0f;
        out[(size_t)o * NPIX + pix] = acc[r] + bv;
    }
}

// -------------------------- proposal decode --------------------------------

__constant__ float c_base_anchors[9][4] = {
    { -84.f,  -40.f,  99.f,  55.f},
    {-176.f,  -88.f, 191.f, 103.f},
    {-360.f, -184.f, 375.f, 191.f},
    { -56.f,  -56.f,  71.f,  71.f},
    {-120.f, -120.f, 135.f, 135.f},
    {-248.f, -248.f, 263.f, 263.f},
    { -36.f,  -80.f,  51.f,  95.f},
    { -80.f, -168.f,  95.f, 183.f},
    {-168.f, -344.f, 183.f, 359.f},
};

__global__ void k_proposals(const float* __restrict__ cls,   // [32][4096]
                            const float* __restrict__ bbx,   // [48][4096]
                            const float* __restrict__ im_info,
                            float* __restrict__ boxes,       // [NANCH][4]
                            float* __restrict__ scores) {    // [NANCH]
    int i = blockIdx.x * blockDim.x + threadIdx.x;
    if (i >= NANCH) return;
    int p = i / A_NUM;
    int a = i - p * A_NUM;
    int y = p >> 6;
    int x = p & 63;

    float c0 = cls[(size_t)a * NPIX + p];
    float c1 = cls[(size_t)(A_NUM + a) * NPIX + p];
    float fg = 1.0f / (1.0f + __expf(c0 - c1));

    float d0 = bbx[(size_t)(4 * a + 0) * NPIX + p];
    float d1 = bbx[(size_t)(4 * a + 1) * NPIX + p];
    float d2 = bbx[(size_t)(4 * a + 2) * NPIX + p];
    float d3 = bbx[(size_t)(4 * a + 3) * NPIX + p];

    float sx = (float)(x * 16), sy = (float)(y * 16);
    float a0 = c_base_anchors[a][0] + sx;
    float a1 = c_base_anchors[a][1] + sy;
    float a2 = c_base_anchors[a][2] + sx;
    float a3 = c_base_anchors[a][3] + sy;

    float aw = a2 - a0 + 1.0f;
    float ah = a3 - a1 + 1.0f;
    float acx = a0 + 0.5f * aw;
    float acy = a1 + 0.5f * ah;
    float pcx = d0 * aw + acx;
    float pcy = d1 * ah + acy;
    float pw = __expf(d2) * aw;
    float ph = __expf(d3) * ah;

    float imH = im_info[0], imW = im_info[1], imS = im_info[2];
    float x1 = fminf(fmaxf(pcx - 0.5f * pw, 0.0f), imW - 1.0f);
    float y1 = fminf(fmaxf(pcy - 0.5f * ph, 0.0f), imH - 1.0f);
    float x2 = fminf(fmaxf(pcx + 0.5f * pw, 0.0f), imW - 1.0f);
    float y2 = fminf(fmaxf(pcy + 0.5f * ph, 0.0f), imH - 1.0f);

    float ms = 16.0f * imS;
    bool keep = (x2 - x1 + 1.0f >= ms) && (y2 - y1 + 1.0f >= ms);

    boxes[i * 4 + 0] = x1;
    boxes[i * 4 + 1] = y1;
    boxes[i * 4 + 2] = x2;
    boxes[i * 4 + 3] = y2;
    scores[i] = keep ? fg : -INFINITY;
}

// ------------------------------- NMS ---------------------------------------

__global__ void __launch_bounds__(1024)
k_nms(const float* __restrict__ boxes, float* __restrict__ scores,
      float* __restrict__ rois) {
    __shared__ float sv[1024];
    __shared__ int   si[1024];
    __shared__ float sb[4];
    int tid = threadIdx.x;
    for (int it = 0; it < POST_NMS; ++it) {
        float best = -INFINITY;
        int bidx = 0;
        for (int i = tid; i < NANCH; i += 1024) {
            float s = scores[i];
            if (s > best) { best = s; bidx = i; }
        }
        sv[tid] = best;
        si[tid] = bidx;
        __syncthreads();
        for (int off = 512; off > 0; off >>= 1) {
            if (tid < off && sv[tid + off] > sv[tid]) {
                sv[tid] = sv[tid + off];
                si[tid] = si[tid + off];
            }
            __syncthreads();
        }
        int j = si[0];
        if (tid == 0) {
            rois[it * 5 + 0] = 0.0f;
            for (int k = 0; k < 4; ++k) {
                float bk = boxes[j * 4 + k];
                rois[it * 5 + 1 + k] = bk;
                sb[k] = bk;
            }
        }
        __syncthreads();
        float bx1 = sb[0], by1 = sb[1], bx2 = sb[2], by2 = sb[3];
        float barea = (bx2 - bx1 + 1.0f) * (by2 - by1 + 1.0f);
        for (int i = tid; i < NANCH; i += 1024) {
            float x1 = boxes[i * 4 + 0], y1 = boxes[i * 4 + 1];
            float x2 = boxes[i * 4 + 2], y2 = boxes[i * 4 + 3];
            float iw = fmaxf(fminf(bx2, x2) - fmaxf(bx1, x1) + 1.0f, 0.0f);
            float ih = fmaxf(fminf(by2, y2) - fmaxf(by1, y1) + 1.0f, 0.0f);
            float inter = iw * ih;
            float iou = inter / (barea + (x2 - x1 + 1.0f) * (y2 - y1 + 1.0f) - inter);
            if (iou > 0.7f) scores[i] = -INFINITY;
        }
        __syncthreads();
    }
}

// ----------------------------- ROI pooling ---------------------------------

__global__ void k_roipool(const float* __restrict__ feat,  // [512][64][64]
                          const float* __restrict__ rois,  // [300][5]
                          float* __restrict__ out) {       // [300][512][7][7]
    int r = blockIdx.x;
    float rx1 = rois[r * 5 + 1] * (1.0f / 16.0f);
    float ry1 = rois[r * 5 + 2] * (1.0f / 16.0f);
    float rx2 = rois[r * 5 + 3] * (1.0f / 16.0f);
    float ry2 = rois[r * 5 + 4] * (1.0f / 16.0f);
    int x1 = (int)roundf(rx1), y1 = (int)roundf(ry1);
    int x2 = (int)roundf(rx2), y2 = (int)roundf(ry2);
    float rw = (float)max(x2 - x1 + 1, 1);
    float rh = (float)max(y2 - y1 + 1, 1);
    float bw = rw / (float)POOL;
    float bh = rh / (float)POOL;

    for (int idx = threadIdx.x; idx < C_IN * POOL * POOL; idx += blockDim.x) {
        int c  = idx / (POOL * POOL);
        int ph = (idx / POOL) % POOL;
        int pw = idx % POOL;
        int hs = min(max((int)floorf((float)ph * bh) + y1, 0), H_IMG);
        int he = min(max((int)ceilf((float)(ph + 1) * bh) + y1, 0), H_IMG);
        int ws = min(max((int)floorf((float)pw * bw) + x1, 0), W_IMG);
        int we = min(max((int)ceilf((float)(pw + 1) * bw) + x1, 0), W_IMG);
        float m = -INFINITY;
        bool any = false;
        int hcap = min(he, hs + MAXK);
        int wcap = min(we, ws + MAXK);
        for (int ih = hs; ih < hcap; ++ih) {
            int ihc = min(max(ih, 0), H_IMG - 1);
            for (int iw = ws; iw < wcap; ++iw) {
                int iwc = min(max(iw, 0), W_IMG - 1);
                float v = feat[(size_t)c * NPIX + ihc * W_IMG + iwc];
                m = fmaxf(m, v);
                any = true;
            }
        }
        out[((size_t)r * C_IN + c) * (POOL * POOL) + ph * POOL + pw] = any ? m : 0.0f;
    }
}

// ------------------------------ launcher -----------------------------------

extern "C" void kernel_launch(void* const* d_in, const int* in_sizes, int n_in,
                              void* d_out, int out_size, void* d_ws, size_t ws_size,
                              hipStream_t stream) {
    const float* features = (const float*)d_in[0];
    const float* im_info  = (const float*)d_in[1];
    const float* rpn_w    = (const float*)d_in[2];
    const float* rpn_b    = (const float*)d_in[3];
    const float* score_w  = (const float*)d_in[4];
    const float* score_b  = (const float*)d_in[5];
    const float* bbox_w   = (const float*)d_in[6];
    const float* bbox_b   = (const float*)d_in[7];

    char* ws = (char*)d_ws;
    // workspace layout (bytes)
    const size_t OFF_XP    = 0;                         // 66*66*512*2  = 4,460,544
    const size_t OFF_WP    = OFF_XP    + 4460544;       // 4,718,592
    const size_t OFF_SP    = OFF_WP    + 4718592;       // 32,768
    const size_t OFF_BP    = OFF_SP    + 32768;         // 49,152
    const size_t OFF_XMID  = OFF_BP    + 49152;         // 4,194,304
    const size_t OFF_CLS   = OFF_XMID  + 4194304;       // 32*4096*4 = 524,288
    const size_t OFF_BBX   = OFF_CLS   + 524288;        // 48*4096*4 = 786,432
    const size_t OFF_BOX   = OFF_BBX   + 786432;        // 589,824
    const size_t OFF_SCR   = OFF_BOX   + 589824;        // 147,456

    unsigned short* xp   = (unsigned short*)(ws + OFF_XP);
    unsigned short* wp   = (unsigned short*)(ws + OFF_WP);
    unsigned short* sp   = (unsigned short*)(ws + OFF_SP);
    unsigned short* bp   = (unsigned short*)(ws + OFF_BP);
    unsigned short* xmid = (unsigned short*)(ws + OFF_XMID);
    float* clsbuf = (float*)(ws + OFF_CLS);
    float* bbxbuf = (float*)(ws + OFF_BBX);
    float* boxes  = (float*)(ws + OFF_BOX);
    float* scores = (float*)(ws + OFF_SCR);

    float* out_pooled = (float*)d_out;
    float* out_rois   = out_pooled + (size_t)POST_NMS * C_IN * POOL * POOL;

    // 1) zero padded activation buffer (border stays zero)
    (void)hipMemsetAsync(xp, 0, 4460544, stream);

    // 2) conversions / repacks
    k_feat_to_bf16<<<(C_IN * NPIX + 255) / 256, 256, 0, stream>>>(features, xp);
    k_repack_conv_w<<<(32 * 9 * 16 * 32 * 16 + 255) / 256, 256, 0, stream>>>(rpn_w, wp);
    k_repack_1x1_w<<<(2 * 16 * 32 * 16 + 255) / 256, 256, 0, stream>>>(score_w, 18, 2, sp);
    k_repack_1x1_w<<<(3 * 16 * 32 * 16 + 255) / 256, 256, 0, stream>>>(bbox_w, 36, 3, bp);

    // 3) 3x3 conv: 512 blocks x 512 threads (16 waves, 64x64 macro tile)
    k_conv3x3_wmma<<<512, 512, 0, stream>>>(xp, wp, rpn_b, xmid);

    // 4) 1x1 heads: score (2 o-tiles) and bbox (3 o-tiles)
    k_gemm1x1_wmma<<<(2 * 256 * 32) / 256, 256, 0, stream>>>(xmid, sp, score_b, 18, clsbuf);
    k_gemm1x1_wmma<<<(3 * 256 * 32) / 256, 256, 0, stream>>>(xmid, bp, bbox_b, 36, bbxbuf);

    // 5) decode proposals
    k_proposals<<<(NANCH + 255) / 256, 256, 0, stream>>>(clsbuf, bbxbuf, im_info, boxes, scores);

    // 6) greedy NMS -> rois (written straight into d_out tail section)
    k_nms<<<1, 1024, 0, stream>>>(boxes, scores, out_rois);

    // 7) ROI max pooling from original f32 features
    k_roipool<<<POST_NMS, 256, 0, stream>>>(features, out_rois, out_pooled);
}
